// LinearLookbackAttention_34144990003565
// MI455X (gfx1250) — matrix operand order repaired and checked by reading, not compiled, not run on hardware
//
#include <hip/hip_runtime.h>
#include <hip/hip_bf16.h>

// ---------------- problem constants ----------------
#define BB     2
#define TT     4096
#define DMODEL 768
#define NH     4
#define DH     64
#define DIMQ   256          // NH*DH
#define CHK    64           // chunk length for chunked linear attention
#define NCHK   (TT / CHK)   // 64 chunks
#define BHN    (BB * NH)    // 8
#define MROWS  (BB * TT)    // 8192

typedef __bf16 bf16;
typedef __attribute__((ext_vector_type(16))) __bf16 v16bf;
typedef __attribute__((ext_vector_type(8)))  __bf16 v8bf;
typedef __attribute__((ext_vector_type(4)))  __bf16 v4bf;
typedef __attribute__((ext_vector_type(8)))  float  v8f;

__device__ __forceinline__ v8f wmma_bf16(v16bf a, v16bf b, v8f c) {
  return __builtin_amdgcn_wmma_f32_16x16x32_bf16(false, a, false, b, (short)0, c,
                                                 false, false);
}

__device__ __forceinline__ v16bf cat8(v8bf lo, v8bf hi) {
  return __builtin_shufflevector(lo, hi, 0, 1, 2, 3, 4, 5, 6, 7, 8, 9, 10, 11,
                                 12, 13, 14, 15);
}

// A fragment from row-major lds[m][k] (ld elems/row, 16B-aligned rows).
// Lane layout: k = kb..kb+7 (VGPR0-3) and kb+16..kb+23 (VGPR4-7), kb = kbase + (lane>=16)*8.
// Both runs are contiguous -> two ds_load_b128.
__device__ __forceinline__ v16bf load_a_vec(const bf16* lds, int ld, int mbase,
                                            int kbase, int lane) {
  int m  = mbase + (lane & 15);
  int kb = kbase + ((lane >> 4) << 3);
  const bf16* p = lds + m * ld + kb;
  v8bf lo = *(const v8bf*)p;
  v8bf hi = *(const v8bf*)(p + 16);
  return cat8(lo, hi);
}
// B fragment from K-contiguous transposed tile ldsT[n][k]:
// lane needs k = kb..kb+7 and kb+16..kb+23 for one n -> two ds_load_b128.
__device__ __forceinline__ v16bf load_bT_vec(const bf16* ldsT, int ldk, int kbase,
                                             int nbase, int lane) {
  int n  = nbase + (lane & 15);
  int kb = kbase + ((lane >> 4) << 3);
  const bf16* p = ldsT + n * ldk + kb;
  v8bf lo = *(const v8bf*)p;
  v8bf hi = *(const v8bf*)(p + 16);
  return cat8(lo, hi);
}

// ---------------- fused GEMM with epilogues ----------------
// MODE 0: elu+1 -> bf16 (B,H,T,D) layout   (q, k feature map)
// MODE 1: identity -> bf16 (B,H,T,D)       (v)
// MODE 2: sigmoid -> f32 (B*T, 768)        (gate)
// MODE 3: * gate -> f32 (B*T, 768)         (final out @ Wo * gate)
template <int MODE, int NDIM, bool ABF16>
__global__ void proj_gemm_kernel(const void* __restrict__ Aptr,
                                 const float* __restrict__ W,
                                 void* __restrict__ dstp,
                                 const float* __restrict__ gate) {
  constexpr int KD = ABF16 ? DIMQ : DMODEL;
  __shared__ bf16 lA[64 * 32];   // [m][k]
  __shared__ bf16 lBT[64 * 32];  // [n][k]  (transposed W tile)
  const int tid = threadIdx.x;
  const int lane = tid & 31, wid = tid >> 5;
  const int m0 = blockIdx.x * 64, n0 = blockIdx.y * 64;
  const int wm = (wid >> 1) * 32, wn = (wid & 1) * 32;
  v8f acc[2][2] = {};
  for (int kk = 0; kk < KD; kk += 32) {
    // ---- A tile 64x32 (vectorized global reads, row-major LDS) ----
#pragma unroll
    for (int i = 0; i < 4; ++i) {
      int chunk = tid + i * 128;           // 0..511 quads
      int r = chunk >> 3;
      int c4 = (chunk & 7) << 2;
      if constexpr (ABF16) {
        const bf16* gp = (const bf16*)Aptr + (size_t)(m0 + r) * KD + kk + c4;
        *(v4bf*)(lA + r * 32 + c4) = *(const v4bf*)gp;
      } else {
        const float* gp = (const float*)Aptr + (size_t)(m0 + r) * KD + kk + c4;
        float4 f = *(const float4*)gp;
        v4bf o;
        o[0] = (bf16)f.x; o[1] = (bf16)f.y; o[2] = (bf16)f.z; o[3] = (bf16)f.w;
        *(v4bf*)(lA + r * 32 + c4) = o;
        if (kk + 32 < KD) __builtin_prefetch(gp + 32, 0, 3);
      }
    }
    // ---- B tile 32x64 -> transposed LDS [n][k] ----
#pragma unroll
    for (int i = 0; i < 4; ++i) {
      int chunk = tid + i * 128;           // 0..511 quads
      int r = chunk >> 4;                  // k row 0..31
      int c4 = (chunk & 15) << 2;          // n 0..60
      const float* gp = W + (size_t)(kk + r) * NDIM + n0 + c4;
      float4 f = *(const float4*)gp;
      lBT[(c4 + 0) * 32 + r] = (bf16)f.x;
      lBT[(c4 + 1) * 32 + r] = (bf16)f.y;
      lBT[(c4 + 2) * 32 + r] = (bf16)f.z;
      lBT[(c4 + 3) * 32 + r] = (bf16)f.w;
      if (kk + 32 < KD) __builtin_prefetch(gp + 32 * NDIM, 0, 3);
    }
    __syncthreads();
    v16bf a0 = load_a_vec(lA, 32, wm, 0, lane);
    v16bf a1 = load_a_vec(lA, 32, wm + 16, 0, lane);
    v16bf b0 = load_bT_vec(lBT, 32, 0, wn, lane);
    v16bf b1 = load_bT_vec(lBT, 32, 0, wn + 16, lane);
    acc[0][0] = wmma_bf16(a0, b0, acc[0][0]);
    acc[0][1] = wmma_bf16(a0, b1, acc[0][1]);
    acc[1][0] = wmma_bf16(a1, b0, acc[1][0]);
    acc[1][1] = wmma_bf16(a1, b1, acc[1][1]);
    __syncthreads();
  }
#pragma unroll
  for (int mi = 0; mi < 2; ++mi)
#pragma unroll
    for (int ni = 0; ni < 2; ++ni)
#pragma unroll
      for (int i = 0; i < 8; ++i) {
        int r = m0 + wm + mi * 16 + i + ((lane >> 4) << 3);
        int c = n0 + wn + ni * 16 + (lane & 15);
        float f = acc[mi][ni][i];
        if constexpr (MODE == 0 || MODE == 1) {
          float val = (MODE == 0) ? ((f > 0.f) ? (f + 1.f) : __expf(f)) : f;
          int b = r >> 12, t = r & (TT - 1);
          int h = c >> 6, d = c & 63;
          ((bf16*)dstp)[(((size_t)(b * NH + h)) * TT + t) * DH + d] = (bf16)val;
        } else if constexpr (MODE == 2) {
          ((float*)dstp)[(size_t)r * DMODEL + c] = 1.f / (1.f + __expf(-f));
        } else {
          ((float*)dstp)[(size_t)r * DMODEL + c] =
              f * gate[(size_t)r * DMODEL + c];
        }
      }
}

// ---------------- per-chunk state: S = V^T K (reference einsum order), z = sum k ----
__global__ void chunk_state_kernel(const bf16* __restrict__ kbf,
                                   const bf16* __restrict__ vbf,
                                   float* __restrict__ Sl, float* __restrict__ zl) {
  __shared__ bf16 lKt[64 * 64];  // [d][t]
  __shared__ bf16 lVt[64 * 64];  // [e][t]
  const int tid = threadIdx.x, lane = tid & 31, wid = tid >> 5;
  const int blk = blockIdx.x, bh = blk / NCHK, ch = blk % NCHK;
  const size_t base = ((size_t)bh * TT + (size_t)ch * CHK) * DH;
  // load 64x64 row-major [t][d] from global (16B vector reads), transpose into LDS
#pragma unroll
  for (int i = 0; i < 4; ++i) {
    int chunk = tid + i * 128;  // 0..511 (8-elem groups)
    int t = chunk >> 3;
    int d8 = (chunk & 7) << 3;
    v8bf kv = *(const v8bf*)(kbf + base + t * 64 + d8);
    v8bf vv = *(const v8bf*)(vbf + base + t * 64 + d8);
#pragma unroll
    for (int j = 0; j < 8; ++j) {
      lKt[(d8 + j) * 64 + t] = kv[j];
      lVt[(d8 + j) * 64 + t] = vv[j];
    }
  }
  __syncthreads();
  const int wm = (wid >> 1) * 32, wn = (wid & 1) * 32;
  v8f acc[2][2] = {};
#pragma unroll
  for (int ks = 0; ks < 64; ks += 32) {
    v16bf a0 = load_a_vec(lVt, 64, wm, ks, lane);        // A[e][t] = V[t][e]
    v16bf a1 = load_a_vec(lVt, 64, wm + 16, ks, lane);
    v16bf b0 = load_bT_vec(lKt, 64, ks, wn, lane);       // B[t][d] = K[t][d]
    v16bf b1 = load_bT_vec(lKt, 64, ks, wn + 16, lane);
    acc[0][0] = wmma_bf16(a0, b0, acc[0][0]);
    acc[0][1] = wmma_bf16(a0, b1, acc[0][1]);
    acc[1][0] = wmma_bf16(a1, b0, acc[1][0]);
    acc[1][1] = wmma_bf16(a1, b1, acc[1][1]);
  }
  const size_t sb = (size_t)blk * 4096;
#pragma unroll
  for (int mi = 0; mi < 2; ++mi)
#pragma unroll
    for (int ni = 0; ni < 2; ++ni)
#pragma unroll
      for (int i = 0; i < 8; ++i) {
        int r = wm + mi * 16 + i + ((lane >> 4) << 3);
        int c = wn + ni * 16 + (lane & 15);
        Sl[sb + r * 64 + c] = acc[mi][ni][i];
      }
  if (tid < 64) {  // z[d] = sum_t K[t][d] : contiguous in lKt
    float s = 0.f;
    for (int t = 0; t < 64; ++t) s += (float)lKt[tid * 64 + t];
    zl[(size_t)blk * 64 + tid] = s;
  }
}

// ---------------- exclusive prefix over chunks (per b,h) ----------------
__global__ void chunk_prefix_kernel(const float* __restrict__ Sl,
                                    const float* __restrict__ zl,
                                    float* __restrict__ Sp, float* __restrict__ zp) {
  const int bh = blockIdx.x, tid = threadIdx.x;
  float run[16];
#pragma unroll
  for (int j = 0; j < 16; ++j) run[j] = 0.f;
  float runz = 0.f;
  for (int c = 0; c < NCHK; ++c) {
    size_t base = ((size_t)bh * NCHK + c) * 4096;
#pragma unroll
    for (int j = 0; j < 16; ++j) {
      int idx = tid + j * 256;
      Sp[base + idx] = run[j];
      run[j] += Sl[base + idx];
    }
    if (tid < 64) {
      size_t zb = ((size_t)bh * NCHK + c) * 64 + tid;
      zp[zb] = runz;
      runz += zl[zb];
    }
  }
}

// ---------------- per-chunk output ----------------
// out[t][d] = Q[t]·Sp[:,d]  +  sum_{s<=t} (q_t·v_s) k[s][d]
// den[t]    = max( q_t·z_pref + sum_{s<=t} q_t·k_s , 1e-6 )
__global__ void attn_chunk_kernel(const bf16* __restrict__ qbf,
                                  const bf16* __restrict__ kbf,
                                  const bf16* __restrict__ vbf,
                                  const float* __restrict__ Sp,
                                  const float* __restrict__ zp,
                                  bf16* __restrict__ attn) {
  __shared__ bf16 lQ[4096];    // [t][e]
  __shared__ bf16 lK[4096];    // [s][e]  (== BT layout for Q·K^T)
  __shared__ bf16 lKt[4096];   // [d][s]  (BT layout for Sc@K)
  __shared__ bf16 lV[4096];    // [s][e]  (== BT layout for Q·V^T)
  __shared__ bf16 lSpT[4096];  // [d][e]  (BT layout for Q@Sp)
  __shared__ bf16 lSb[4096];   // masked score buffer [t][s]
  __shared__ float lden[64], lz[64];
  const int tid = threadIdx.x, lane = tid & 31, wid = tid >> 5;
  const int blk = blockIdx.x, bh = blk / NCHK, ch = blk % NCHK;
  const int bb = bh >> 2, hh = bh & 3;
  const size_t base = ((size_t)bh * TT + (size_t)ch * CHK) * DH;
#pragma unroll
  for (int i = 0; i < 4; ++i) {
    int chunk = tid + i * 128;  // 0..511 (8-elem groups)
    int t = chunk >> 3;
    int d8 = (chunk & 7) << 3;
    *(v8bf*)(lQ + t * 64 + d8) = *(const v8bf*)(qbf + base + t * 64 + d8);
    *(v8bf*)(lV + t * 64 + d8) = *(const v8bf*)(vbf + base + t * 64 + d8);
    v8bf kv = *(const v8bf*)(kbf + base + t * 64 + d8);
    *(v8bf*)(lK + t * 64 + d8) = kv;
#pragma unroll
    for (int j = 0; j < 8; ++j) lKt[(d8 + j) * 64 + t] = kv[j];
  }
#pragma unroll
  for (int i = 0; i < 8; ++i) {  // S_prefix f32 [e][d] -> bf16 transposed [d][e]
    int chunk = tid + i * 128;   // 0..1023 quads
    int e = chunk >> 4;
    int d4 = (chunk & 15) << 2;
    float4 f = *(const float4*)(Sp + (size_t)blk * 4096 + e * 64 + d4);
    lSpT[(d4 + 0) * 64 + e] = (bf16)f.x;
    lSpT[(d4 + 1) * 64 + e] = (bf16)f.y;
    lSpT[(d4 + 2) * 64 + e] = (bf16)f.z;
    lSpT[(d4 + 3) * 64 + e] = (bf16)f.w;
  }
  if (tid < 64) lz[tid] = zp[(size_t)blk * 64 + tid];
  __syncthreads();
  const int wm = (wid >> 1) * 32, wn = (wid & 1) * 32;

  // inter-chunk contribution: acc = Q @ Sp
  v8f acc[2][2] = {};
#pragma unroll
  for (int ks = 0; ks < 64; ks += 32) {
    v16bf a0 = load_a_vec(lQ, 64, wm, ks, lane);
    v16bf a1 = load_a_vec(lQ, 64, wm + 16, ks, lane);
    v16bf b0 = load_bT_vec(lSpT, 64, ks, wn, lane);
    v16bf b1 = load_bT_vec(lSpT, 64, ks, wn + 16, lane);
    acc[0][0] = wmma_bf16(a0, b0, acc[0][0]);
    acc[0][1] = wmma_bf16(a0, b1, acc[0][1]);
    acc[1][0] = wmma_bf16(a1, b0, acc[1][0]);
    acc[1][1] = wmma_bf16(a1, b1, acc[1][1]);
  }
  // denominator intra scores: q_t · k_s, masked to bf16 buffer
  {
    v8f sck[2][2] = {};
#pragma unroll
    for (int ks = 0; ks < 64; ks += 32) {
      v16bf a0 = load_a_vec(lQ, 64, wm, ks, lane);
      v16bf a1 = load_a_vec(lQ, 64, wm + 16, ks, lane);
      v16bf b0 = load_bT_vec(lK, 64, ks, wn, lane);
      v16bf b1 = load_bT_vec(lK, 64, ks, wn + 16, lane);
      sck[0][0] = wmma_bf16(a0, b0, sck[0][0]);
      sck[0][1] = wmma_bf16(a0, b1, sck[0][1]);
      sck[1][0] = wmma_bf16(a1, b0, sck[1][0]);
      sck[1][1] = wmma_bf16(a1, b1, sck[1][1]);
    }
#pragma unroll
    for (int mi = 0; mi < 2; ++mi)
#pragma unroll
      for (int ni = 0; ni < 2; ++ni)
#pragma unroll
        for (int i = 0; i < 8; ++i) {
          int r = wm + mi * 16 + i + ((lane >> 4) << 3);
          int c = wn + ni * 16 + (lane & 15);
          lSb[r * 64 + c] = (bf16)((c <= r) ? sck[mi][ni][i] : 0.f);
        }
  }
  __syncthreads();
  if (tid < 64) {
    int r = tid;
    float qz = 0.f;
    for (int d = 0; d < 64; ++d) qz += (float)lQ[r * 64 + d] * lz[d];
    float s = 0.f;
    for (int j = 0; j < 64; ++j) s += (float)lSb[r * 64 + j];  // pre-masked
    lden[r] = fmaxf(qz + s, 1e-6f);
  }
  __syncthreads();
  // intra scores per reference index order: q_t · v_s, masked bf16
  {
    v8f scv[2][2] = {};
#pragma unroll
    for (int ks = 0; ks < 64; ks += 32) {
      v16bf a0 = load_a_vec(lQ, 64, wm, ks, lane);
      v16bf a1 = load_a_vec(lQ, 64, wm + 16, ks, lane);
      v16bf b0 = load_bT_vec(lV, 64, ks, wn, lane);
      v16bf b1 = load_bT_vec(lV, 64, ks, wn + 16, lane);
      scv[0][0] = wmma_bf16(a0, b0, scv[0][0]);
      scv[0][1] = wmma_bf16(a0, b1, scv[0][1]);
      scv[1][0] = wmma_bf16(a1, b0, scv[1][0]);
      scv[1][1] = wmma_bf16(a1, b1, scv[1][1]);
    }
    __syncthreads();  // lden reads of lSb complete before overwrite
#pragma unroll
    for (int mi = 0; mi < 2; ++mi)
#pragma unroll
      for (int ni = 0; ni < 2; ++ni)
#pragma unroll
        for (int i = 0; i < 8; ++i) {
          int r = wm + mi * 16 + i + ((lane >> 4) << 3);
          int c = wn + ni * 16 + (lane & 15);
          lSb[r * 64 + c] = (bf16)((c <= r) ? scv[mi][ni][i] : 0.f);
        }
  }
  __syncthreads();
  // intra output: acc += Sb @ K
#pragma unroll
  for (int ks = 0; ks < 64; ks += 32) {
    v16bf a0 = load_a_vec(lSb, 64, wm, ks, lane);
    v16bf a1 = load_a_vec(lSb, 64, wm + 16, ks, lane);
    v16bf b0 = load_bT_vec(lKt, 64, ks, wn, lane);
    v16bf b1 = load_bT_vec(lKt, 64, ks, wn + 16, lane);
    acc[0][0] = wmma_bf16(a0, b0, acc[0][0]);
    acc[0][1] = wmma_bf16(a0, b1, acc[0][1]);
    acc[1][0] = wmma_bf16(a1, b0, acc[1][0]);
    acc[1][1] = wmma_bf16(a1, b1, acc[1][1]);
  }
  const int t0 = ch * CHK;
#pragma unroll
  for (int mi = 0; mi < 2; ++mi)
#pragma unroll
    for (int ni = 0; ni < 2; ++ni)
#pragma unroll
      for (int i = 0; i < 8; ++i) {
        int r = wm + mi * 16 + i + ((lane >> 4) << 3);
        int c = wn + ni * 16 + (lane & 15);
        float val = acc[mi][ni][i] / lden[r];
        attn[((size_t)(bb * TT + t0 + r)) * DIMQ + hh * DH + c] = (bf16)val;
      }
}

// ---------------- host launcher ----------------
extern "C" void kernel_launch(void* const* d_in, const int* in_sizes, int n_in,
                              void* d_out, int out_size, void* d_ws, size_t ws_size,
                              hipStream_t stream) {
  (void)in_sizes; (void)n_in; (void)out_size; (void)ws_size;
  const float* x  = (const float*)d_in[0];
  const float* Wq = (const float*)d_in[1];
  const float* Wk = (const float*)d_in[2];
  const float* Wv = (const float*)d_in[3];
  const float* Wo = (const float*)d_in[4];
  const float* Wg = (const float*)d_in[5];

  char* ws = (char*)d_ws;
  size_t off = 0;
  bf16* qbf = (bf16*)(ws + off); off += (size_t)BHN * TT * DH * sizeof(bf16);
  bf16* kbf = (bf16*)(ws + off); off += (size_t)BHN * TT * DH * sizeof(bf16);
  bf16* vbf = (bf16*)(ws + off); off += (size_t)BHN * TT * DH * sizeof(bf16);
  float* gate = (float*)(ws + off); off += (size_t)MROWS * DMODEL * sizeof(float);
  bf16* attn = (bf16*)(ws + off); off += (size_t)MROWS * DIMQ * sizeof(bf16);
  float* Sl = (float*)(ws + off); off += (size_t)BHN * NCHK * DH * DH * sizeof(float);
  float* zl = (float*)(ws + off); off += (size_t)BHN * NCHK * DH * sizeof(float);
  float* Sp = (float*)(ws + off); off += (size_t)BHN * NCHK * DH * DH * sizeof(float);
  float* zp = (float*)(ws + off); off += (size_t)BHN * NCHK * DH * sizeof(float);

  dim3 blk(128);
  proj_gemm_kernel<0, DIMQ, false>
      <<<dim3(MROWS / 64, DIMQ / 64), blk, 0, stream>>>(x, Wq, qbf, nullptr);
  proj_gemm_kernel<0, DIMQ, false>
      <<<dim3(MROWS / 64, DIMQ / 64), blk, 0, stream>>>(x, Wk, kbf, nullptr);
  proj_gemm_kernel<1, DIMQ, false>
      <<<dim3(MROWS / 64, DIMQ / 64), blk, 0, stream>>>(x, Wv, vbf, nullptr);
  proj_gemm_kernel<2, DMODEL, false>
      <<<dim3(MROWS / 64, DMODEL / 64), blk, 0, stream>>>(x, Wg, gate, nullptr);
  chunk_state_kernel<<<BHN * NCHK, 128, 0, stream>>>(kbf, vbf, Sl, zl);
  chunk_prefix_kernel<<<BHN, 256, 0, stream>>>(Sl, zl, Sp, zp);
  attn_chunk_kernel<<<BHN * NCHK, 128, 0, stream>>>(qbf, kbf, vbf, Sp, zp, attn);
  proj_gemm_kernel<3, DMODEL, true>
      <<<dim3(MROWS / 64, DMODEL / 64), blk, 0, stream>>>(attn, Wo, d_out, gate);
}
